// MultiHGDModule_1632_87540023427895
// MI455X (gfx1250) — matrix operand order repaired and checked by reading, not compiled, not run on hardware
//
#include <hip/hip_runtime.h>
#include <hip/hip_bf16.h>

// ---------------------------------------------------------------------------
// MultiHGD on MI455X (gfx1250, wave32, WMMA + Tensor Data Mover).
// All 1x1 convs / einsums are bf16 WMMA GEMMs; the shared activation (B)
// operand is staged into LDS by TDM (tensor_load_to_lds), double buffered,
// and consumed by 8 waves (128 M-rows) per block.
// Activations: bf16 pixel-major [n][p][c]; weights: bf16 [O][K] row-major.
// Scratch layout (requires ws_size >= ~150 MB):
//   [0,4MB)      weight bf16 pool
//   [4,12MB)     xcat16   [8][256][1536]
//   [12,14MB)    fcat f32 [8][256][256]   (channel-major, for pack/va)
//   [14,16MB)    fcen f32 [8][256][256]   (channel-major, for softmax)
//   [16,17MB)    fcat16   [8][256][256]
//   [17,18MB)    fnorm16  [8][256][256]
//   [18MB)       va[8][256] f32 ; +64KB: bprime[8][256] f32
//   [19,20MB)    xc16     [8][256(o)][256(c)]
//   [20,116MB)   gcat16   [8][4096][1536]  (NHWC concat @64x64)
//      reused after gc GEMMs: t64@20MB, aff_l@36MB, aff_d@52MB,
//                             xup_l@68MB, xup_d@84MB (16MB each)
//   [116,132MB)  gc_lane16 [8][4096][256]
//   [132,148MB)  gc_driv16 [8][4096][256]
// ---------------------------------------------------------------------------

typedef __bf16 bf16;
typedef bf16         bf16x8   __attribute__((ext_vector_type(8)));
typedef bf16         bf16x16  __attribute__((ext_vector_type(16)));
typedef float        f32x8    __attribute__((ext_vector_type(8)));
typedef unsigned int u32x4    __attribute__((ext_vector_type(4)));
typedef int          i32x4    __attribute__((ext_vector_type(4)));
typedef int          i32x8    __attribute__((ext_vector_type(8)));

#define NB   8
#define CCH  512
#define K3   1536
#define OC   256
#define P0   256
#define P2   4096

// K staged per TDM transfer (bf16 channels); one LDS row = KB*2 = 512B data.
#define KB        256
// LDS row stride: 512B data + 16B TDM pad -> 528B (=264 bf16). Spreads the
// 16-lane b128 read pattern over all 64 LDS banks (2-way = minimum).
#define ROWE      264
#define BUF_ELEMS (64 * ROWE)            // one buffer: 64 pixel rows
#define BUF_BYTES (BUF_ELEMS * 2)        // 33792 B

// ---------------------------------------------------------------------------
// TDM issue: move a [64 px][KB ch] bf16 tile (row stride = ld elems in global)
// into LDS at byte offset lds_off, with 16B pad per 512B row.
// D# packing per CDNA5 ISA 8.3/8.4 (data_size=3 => 8-byte units).
// ---------------------------------------------------------------------------
__device__ __forceinline__ void tdm_load_tile(const bf16* src, int ld, unsigned lds_off) {
  unsigned long long ga = (unsigned long long)(uintptr_t)src;
  u32x4 g0;
  g0[0] = 1u;                                   // count=1, user descriptor
  g0[1] = lds_off;                              // lds_addr (bytes)
  g0[2] = (unsigned)ga;                         // global_addr[31:0]
  g0[3] = (unsigned)(ga >> 32) | 0x80000000u;   // global_addr[56:32] | type=2
  const unsigned tile0   = (KB * 2) / 8;        // 64  (8B units per row)
  const unsigned tile1   = 64;                  // rows (pixels)
  const unsigned stride0 = (unsigned)(ld / 4);  // global row stride, 8B units
  const unsigned tdim    = 0x40000000u;         // huge: never OOB-zeroed
  i32x8 g1;
  g1[0] = (3 << 16)                             // data_size = 8B
        | (1 << 20)                             // pad_enable
        | (6 << 22)                             // pad_interval: 128 DW = 512B
        | (3 << 25);                            // pad_amount: 4 DW = 16B
  g1[1] = (int)((tdim & 0xFFFFu) << 16);                      // tensor_dim0 lo
  g1[2] = (int)((tdim >> 16) | ((tdim & 0xFFFFu) << 16));     // d0 hi | d1 lo
  g1[3] = (int)((tdim >> 16) | (tile0 << 16));                // d1 hi | tile0
  g1[4] = (int)tile1;                                         // tile1 | tile2=0
  g1[5] = (int)stride0;                                       // dim0_stride lo
  g1[6] = 0;                                                  // d0s hi | d1s lo
  g1[7] = 0;
  i32x4 gz4 = {0, 0, 0, 0};
  i32x8 gz8 = {0, 0, 0, 0, 0, 0, 0, 0};
  // 6-arg form (clang-23 / therock-10.0 HIP headers)
  __builtin_amdgcn_tensor_load_to_lds(g0, g1, gz4, gz4, gz8, 0);
}

// ---------------------------------------------------------------------------
// bf16 WMMA GEMM, TDM/LDS-staged B operand.
//   D[m,p] = act( scale*(sum_k A[m,k]*X[p,k]) + bias )
// Block: 8 waves -> 128(M) x 64(N=p) tile; wave w owns rows [mg*128+16w, +16).
// A: [M][lda] bf16 (per-batch stride a_bstride; 0 for shared weights)
// X: [NB][P][ldx] bf16 pixel-major; optional split-K second source X2
//    (K multiple of KB; split boundary must be a KB multiple).
// Outputs: Y16 pixel-major bf16 (+channel offset co) and/or Yf channel-major
//          f32 [NB][M][P] (softmax rows / final NCHW output).
// ---------------------------------------------------------------------------
__global__ __launch_bounds__(256) void gemm_bf16_wmma(
    const bf16* __restrict__ A, long long a_bstride, int lda,
    const bf16* __restrict__ X, int ldx,
    const bf16* __restrict__ X2, int ldx2, int ksplit,
    const float* __restrict__ sc, const float* __restrict__ bi,
    const float* __restrict__ bi_nm, float alpha, int relu,
    bf16* __restrict__ Y16, int ldo, int co,
    float* __restrict__ Yf,
    int M, int K, int P)
{
  // Single static LDS allocation -> assumed at LDS offset 0 (TDM descriptor
  // addresses LDS by raw byte offset).
  __shared__ __align__(16) bf16 smem[2 * BUF_ELEMS];

  const int lane = threadIdx.x & 31;
  const int wid  = threadIdx.x >> 5;            // 0..7
  const int pgroups = P >> 6;
  const int mgroups = M >> 7;                   // M multiple of 128
  int b = blockIdx.x;
  const int pg = b % pgroups;  b /= pgroups;
  const int mg = b % mgroups;
  const int n  = b / mgroups;

  const int l15 = lane & 15;
  const int hi  = lane >> 4;
  const int m0  = (mg << 7) + (wid << 4);
  const int p0  = pg << 6;

  // Visible LDS stores: the TDM builtin's LDS writes are invisible to alias
  // analysis; without a real store the smem loads get undef-folded and
  // hoisted (observed in round-3 asm). TDM overwrites these bytes afterwards,
  // and the barrier orders the init ahead of the first DMA.
  smem[threadIdx.x] = (bf16)0.0f;
  smem[BUF_ELEMS + threadIdx.x] = (bf16)0.0f;
  __syncthreads();

  const bf16* __restrict__ arow =
      A + (long long)n * a_bstride + (long long)(m0 + l15) * lda + hi * 8;

  f32x8 acc[4];
  f32x8 zv = {0.f,0.f,0.f,0.f,0.f,0.f,0.f,0.f};
  acc[0] = zv; acc[1] = zv; acc[2] = zv; acc[3] = zv;

  const int nkb = K / KB;

  // Source row selector for split-K (uniform per kb chunk).
  auto srcptr = [&](int kb) -> const bf16* {
    int kg = kb * KB;
    if (X2 && kg >= ksplit)
      return X2 + ((long long)n * P + p0) * ldx2 + (kg - ksplit);
    return X + ((long long)n * P + p0) * ldx + kg;
  };
  auto srcld = [&](int kb) -> int {
    return (X2 && kb * KB >= ksplit) ? ldx2 : ldx;
  };

  if (wid == 0) tdm_load_tile(srcptr(0), srcld(0), 0u);

  for (int kb = 0; kb < nkb; ++kb) {
    if (wid == 0) {
      if (kb + 1 < nkb) {
        tdm_load_tile(srcptr(kb + 1), srcld(kb + 1), (unsigned)(((kb + 1) & 1) * BUF_BYTES));
        __builtin_amdgcn_s_wait_tensorcnt((short)1);   // buffer kb ready
      } else {
        __builtin_amdgcn_s_wait_tensorcnt((short)0);
      }
    }
    __syncthreads();                                   // publish buffer kb
    asm volatile("" ::: "memory");   // LDS contents changed by TDM: no
                                     // forwarding/hoisting across this point

    const bf16* __restrict__ btile = smem + (kb & 1) * BUF_ELEMS;
    const int kbase = kb * KB;
#pragma unroll 4
    for (int ks = 0; ks < KB; ks += 32) {
      // A fragment (global): lane m=l15, chunks [k0,k0+8) and [k0+16,k0+24)
      bf16x16 a;
      ((bf16x8*)&a)[0] = *(const bf16x8*)(arow + kbase + ks);
      ((bf16x8*)&a)[1] = *(const bf16x8*)(arow + kbase + ks + 16);
      __builtin_prefetch(arow + kbase + ks + 32, 0, 1);

      // B fragments (LDS): lane col=l15(+16t), chunks [k0+8h,+8) and [k0+16+8h,+8)
#pragma unroll
      for (int t = 0; t < 4; ++t) {
        const bf16* bp = btile + (size_t)(t * 16 + l15) * ROWE + ks + hi * 8;
        bf16x16 bfrag;
        ((bf16x8*)&bfrag)[0] = *(const bf16x8*)(bp);      // ds_load_b128
        ((bf16x8*)&bfrag)[1] = *(const bf16x8*)(bp + 16); // ds_load_b128
        acc[t] = __builtin_amdgcn_wmma_f32_16x16x32_bf16(
            false, a, false, bfrag, (short)0, acc[t], false, false);
      }
    }
    asm volatile("" ::: "memory");   // keep LDS reads above the reuse barrier
    __syncthreads();   // all waves done with buffer kb before TDM reuses it
  }

  // Epilogue: lane holds col p, rows m = m0 + hi*8 + r (r=0..7, contiguous).
  const int mo = m0 + (hi << 3);
  float sr[8], br[8];
#pragma unroll
  for (int r = 0; r < 8; ++r) {
    if (sc) {
      sr[r] = sc[mo + r];
      br[r] = bi_nm ? bi_nm[n * M + mo + r] : bi[mo + r];
    } else { sr[r] = alpha; br[r] = 0.0f; }
  }
#pragma unroll
  for (int t = 0; t < 4; ++t) {
    const int p = p0 + t * 16 + l15;
    float v[8];
#pragma unroll
    for (int r = 0; r < 8; ++r) {
      float u = acc[t][r] * sr[r] + br[r];
      if (relu) u = fmaxf(u, 0.0f);
      v[r] = u;
    }
    if (Y16) {
      bf16x8 o;
#pragma unroll
      for (int r = 0; r < 8; ++r) o[r] = (bf16)v[r];
      *(bf16x8*)(Y16 + ((long long)n * P + p) * ldo + co + mo) = o;  // 16B store
    }
    if (Yf) {
#pragma unroll
      for (int r = 0; r < 8; ++r)
        Yf[((long long)n * M + mo + r) * P + p] = v[r];
    }
  }
}

// ---------------------------------------------------------------------------
// Helpers
// ---------------------------------------------------------------------------
__device__ __forceinline__ float bilerp_ac(const float* __restrict__ img, int S,
                                           float sy, float sx) {
  int y0 = (int)sy, x0 = (int)sx;
  float fy = sy - (float)y0, fx = sx - (float)x0;
  int y1 = min(y0 + 1, S - 1), x1 = min(x0 + 1, S - 1);
  float a = img[y0 * S + x0], b = img[y0 * S + x1];
  float c = img[y1 * S + x0], d = img[y1 * S + x1];
  float top = a + (b - a) * fx, bot = c + (d - c) * fx;
  return top + (bot - top) * fy;
}

__global__ void f32_to_bf16_k(const float* __restrict__ s, bf16* __restrict__ d, int n) {
  int i = blockIdx.x * blockDim.x + threadIdx.x;
  if (i < n) d[i] = (bf16)s[i];
}

// xcat16 [n][p(16x16)][c(1536)]: c0-511 g2_down(64->16), 512-1023 g1_down(32->16), 1024-1535 x
__global__ void build_xcat_k(const float* __restrict__ x, const float* __restrict__ g1,
                             const float* __restrict__ g2, bf16* __restrict__ out) {
  int idx = blockIdx.x * 256 + threadIdx.x;          // 8*256*1536 total
  int c = idx % K3; int t = idx / K3;
  int p = t % P0;   int n = t / P0;
  int y = p >> 4, xx = p & 15;
  int sec = c >> 9, cc = c & 511;
  float v;
  if (sec == 2) {
    v = x[(((long long)(n * CCH + cc)) << 8) + (y << 4) + xx];
  } else if (sec == 0) {
    const float* img = g2 + (((long long)(n * CCH + cc)) << 12);
    v = bilerp_ac(img, 64, y * (63.0f / 15.0f), xx * (63.0f / 15.0f));
  } else {
    const float* img = g1 + (((long long)(n * CCH + cc)) << 10);
    v = bilerp_ac(img, 32, y * (31.0f / 15.0f), xx * (31.0f / 15.0f));
  }
  out[idx] = (bf16)v;
}

// gcat16 [n][p(64x64)][c(1536)]: c0-511 guide2, 512-1023 x_up1(32->64), 1024-1535 x_up0(16->64)
__global__ void build_gcat_k(const float* __restrict__ x, const float* __restrict__ g1,
                             const float* __restrict__ g2, bf16* __restrict__ out) {
  int idx = blockIdx.x * 256 + threadIdx.x;          // 8*4096*1536 total
  int c = idx % K3; long long t = idx / K3;
  int p = (int)(t % P2); int n = (int)(t / P2);
  int y = p >> 6, xx = p & 63;
  int sec = c >> 9, cc = c & 511;
  float v;
  if (sec == 0) {
    v = g2[(((long long)(n * CCH + cc)) << 12) + p];
  } else if (sec == 1) {
    const float* img = g1 + (((long long)(n * CCH + cc)) << 10);
    v = bilerp_ac(img, 32, y * (31.0f / 63.0f), xx * (31.0f / 63.0f));
  } else {
    const float* img = x + (((long long)(n * CCH + cc)) << 8);
    v = bilerp_ac(img, 16, y * (15.0f / 63.0f), xx * (15.0f / 63.0f));
  }
  out[idx] = (bf16)v;
}

// f_cat f32 [row=n*256+o][256p] -> bf16 copy + row mean (value_avg)
__global__ __launch_bounds__(256) void pack_fcat_k(const float* __restrict__ f,
                                                   bf16* __restrict__ f16,
                                                   float* __restrict__ va) {
  __shared__ float red[256];
  int row = blockIdx.x, tid = threadIdx.x;
  float v = f[row * 256 + tid];
  f16[row * 256 + tid] = (bf16)v;
  red[tid] = v; __syncthreads();
  for (int s = 128; s > 0; s >>= 1) { if (tid < s) red[tid] += red[tid + s]; __syncthreads(); }
  if (tid == 0) va[row] = red[0] * (1.0f / 256.0f);
}

// softmax over p (256) per (n,c) row; f32 in, bf16 out
__global__ __launch_bounds__(256) void softmax_row_k(const float* __restrict__ f,
                                                     bf16* __restrict__ o16) {
  __shared__ float red[256];
  int row = blockIdx.x, tid = threadIdx.x;
  float v = f[row * 256 + tid];
  red[tid] = v; __syncthreads();
  for (int s = 128; s > 0; s >>= 1) { if (tid < s) red[tid] = fmaxf(red[tid], red[tid + s]); __syncthreads(); }
  float m = red[0]; __syncthreads();
  float e = __expf(v - m);
  red[tid] = e; __syncthreads();
  for (int s = 128; s > 0; s >>= 1) { if (tid < s) red[tid] += red[tid + s]; __syncthreads(); }
  o16[row * 256 + tid] = (bf16)(e / red[0]);
}

// b'[n][m] = b_aff1[m] + s_aff1[m] * sum_k w_aff1[m][k]*va[n][k]
// (folds value_avg into the aff1 bias -> no z=gc+va intermediate needed)
__global__ __launch_bounds__(256) void bprime_k(const float* __restrict__ w1,
                                                const float* __restrict__ s1,
                                                const float* __restrict__ b1,
                                                const float* __restrict__ va,
                                                float* __restrict__ bp) {
  int n = blockIdx.x, m = threadIdx.x;
  const float* wr = w1 + m * 256;
  const float* vv = va + n * 256;
  float sum = 0.0f;
  for (int k = 0; k < 256; ++k) sum += wr[k] * vv[k];
  bp[n * 256 + m] = b1[m] + s1[m] * sum;
}

// ---------------------------------------------------------------------------
extern "C" void kernel_launch(void* const* d_in, const int* in_sizes, int n_in,
                              void* d_out, int out_size, void* d_ws, size_t ws_size,
                              hipStream_t stream) {
  const float* x      = (const float*)d_in[0];
  const float* g1     = (const float*)d_in[1];
  const float* g2     = (const float*)d_in[2];
  const float* w_cat  = (const float*)d_in[3];
  const float* s_cat  = (const float*)d_in[4];
  const float* b_cat  = (const float*)d_in[5];
  const float* w_cen1 = (const float*)d_in[6];
  const float* s_cen1 = (const float*)d_in[7];
  const float* b_cen1 = (const float*)d_in[8];
  const float* w_cen2 = (const float*)d_in[9];
  const float* s_cen2 = (const float*)d_in[10];
  const float* b_cen2 = (const float*)d_in[11];
  const float* w_lane = (const float*)d_in[12];
  const float* s_lane = (const float*)d_in[13];
  const float* b_lane = (const float*)d_in[14];
  const float* w_driv = (const float*)d_in[15];
  const float* s_driv = (const float*)d_in[16];
  const float* b_driv = (const float*)d_in[17];
  const float* w_aff1 = (const float*)d_in[18];
  const float* s_aff1 = (const float*)d_in[19];
  const float* b_aff1 = (const float*)d_in[20];
  const float* w_aff2 = (const float*)d_in[21];
  const float* s_aff2 = (const float*)d_in[22];
  const float* b_aff2 = (const float*)d_in[23];
  const float* w_up   = (const float*)d_in[24];
  const float* s_up   = (const float*)d_in[25];
  const float* b_up   = (const float*)d_in[26];
  float* out = (float*)d_out;
  char*  ws  = (char*)d_ws;

  // ---- scratch carving (see header comment) ----
  bf16* wb       = (bf16*)ws;
  bf16* wcat16   = wb;
  bf16* wcen1_16 = wb + 393216;
  bf16* wcen2_16 = wb + 786432;
  bf16* wlane16  = wb + 851968;
  bf16* wdriv16  = wb + 1245184;
  bf16* waff1_16 = wb + 1507328;
  bf16* waff2_16 = wb + 1572864;
  bf16* wup16    = wb + 1638400;

  bf16*  xcat16  = (bf16*) (ws + (4LL  << 20));
  float* fcatf   = (float*)(ws + (12LL << 20));
  float* fcenf   = (float*)(ws + (14LL << 20));
  bf16*  fcat16  = (bf16*) (ws + (16LL << 20));
  bf16*  fnorm16 = (bf16*) (ws + (17LL << 20));
  float* va      = (float*)(ws + (18LL << 20));
  float* bprime  = (float*)(ws + (18LL << 20) + 65536);
  bf16*  xc16    = (bf16*) (ws + (19LL << 20));
  bf16*  gcat    = (bf16*) (ws + (20LL << 20));
  bf16*  t64     = (bf16*) (ws + (20LL << 20));   // reuses gcat region
  bf16*  aff_l   = (bf16*) (ws + (36LL << 20));
  bf16*  aff_d   = (bf16*) (ws + (52LL << 20));
  bf16*  xup_l   = (bf16*) (ws + (68LL << 20));
  bf16*  xup_d   = (bf16*) (ws + (84LL << 20));
  bf16*  gc_l    = (bf16*) (ws + (116LL << 20));
  bf16*  gc_d    = (bf16*) (ws + (132LL << 20));

  auto cvt = [&](const float* s, bf16* d, int n) {
    f32_to_bf16_k<<<(n + 255) / 256, 256, 0, stream>>>(s, d, n);
  };
  auto gemm = [&](const bf16* A, long long abst, int lda, const bf16* X, int ldx,
                  const bf16* X2, int ldx2, int ksplit,
                  const float* sc, const float* bi, const float* binm,
                  float alpha, int relu,
                  bf16* Y16, int ldo, int co, float* Yf, int M, int K, int P) {
    int blocks = NB * (M / 128) * (P / 64);
    gemm_bf16_wmma<<<blocks, 256, 0, stream>>>(A, abst, lda, X, ldx, X2, ldx2, ksplit,
                                               sc, bi, binm, alpha, relu,
                                               Y16, ldo, co, Yf, M, K, P);
  };

  // ---- weight down-conversion (once) ----
  cvt(w_cat,  wcat16,   393216);
  cvt(w_cen1, wcen1_16, 393216);
  cvt(w_cen2, wcen2_16, 65536);
  cvt(w_lane, wlane16,  393216);
  cvt(w_driv, wdriv16,  262144);
  cvt(w_aff1, waff1_16, 65536);
  cvt(w_aff2, waff2_16, 65536);
  cvt(w_up,   wup16,    131072);

  // ---- Phase A: 16x16 resolution ----
  build_xcat_k<<<(NB * P0 * K3) / 256, 256, 0, stream>>>(x, g1, g2, xcat16);

  // f_cat = relu(bn(w_cat @ xcat))  -> f32 channel-major
  gemm(wcat16, 0, K3, xcat16, K3, nullptr, 0, K3,
       s_cat, b_cat, nullptr, 1.0f, 1, nullptr, 0, 0, fcatf, OC, K3, P0);
  pack_fcat_k<<<NB * OC, 256, 0, stream>>>(fcatf, fcat16, va);

  // f_center = bn2(w_cen2 @ relu(bn1(w_cen1 @ xcat)))
  gemm(wcen1_16, 0, K3, xcat16, K3, nullptr, 0, K3,
       s_cen1, b_cen1, nullptr, 1.0f, 1, t64, OC, 0, nullptr, OC, K3, P0);
  gemm(wcen2_16, 0, OC, t64, OC, nullptr, 0, OC,
       s_cen2, b_cen2, nullptr, 1.0f, 0, nullptr, 0, 0, fcenf, OC, OC, P0);
  softmax_row_k<<<NB * OC, 256, 0, stream>>>(fcenf, fnorm16);
  bprime_k<<<NB, 256, 0, stream>>>(w_aff1, s_aff1, b_aff1, va, bprime);

  // x_center^T: xc[n][o][c] = (1/16) * sum_p fnorm[n,c,p]*fcat[n,o,p]
  //   A = fnorm (per-batch, M=c), B cols = o (fcat16 [n][o][p]), D -> xc16 [n][o][c]
  gemm(fnorm16, 65536, OC, fcat16, OC, nullptr, 0, OC,
       nullptr, nullptr, nullptr, 0.0625f, 0, xc16, OC, 0, nullptr, OC, OC, P0);

  // ---- Phase B: 64x64 resolution ----
  build_gcat_k<<<(NB * P2 * K3) / 256, 256, 0, stream>>>(x, g1, g2, gcat);

  // gc_lane = relu(bn(w_lane @ gcat));   gc_driv uses channels 512..1535 of gcat
  gemm(wlane16, 0, K3, gcat, K3, nullptr, 0, K3,
       s_lane, b_lane, nullptr, 1.0f, 1, gc_l, OC, 0, nullptr, OC, K3, P2);
  gemm(wdriv16, 0, 1024, gcat + 512, K3, nullptr, 0, 1024,
       s_driv, b_driv, nullptr, 1.0f, 1, gc_d, OC, 0, nullptr, OC, 1024, P2);

  const long long out_half = (long long)NB * OC * P2;

  // ---- lane branch ----
  gemm(waff1_16, 0, OC, gc_l, OC, nullptr, 0, OC,
       s_aff1, b_aff1, bprime, 1.0f, 1, t64, OC, 0, nullptr, OC, OC, P2);
  gemm(waff2_16, 0, OC, t64, OC, nullptr, 0, OC,
       s_aff2, b_aff2, nullptr, 1.0f, 1, aff_l, OC, 0, nullptr, OC, OC, P2);
  gemm(xc16, 65536, OC, aff_l, OC, nullptr, 0, OC,
       nullptr, nullptr, nullptr, 1.0f, 0, xup_l, OC, 0, nullptr, OC, OC, P2);
  gemm(wup16, 0, 512, xup_l, OC, gc_l, OC, 256,
       s_up, b_up, nullptr, 1.0f, 1, nullptr, 0, 0, out, OC, 512, P2);

  // ---- driv branch ----
  gemm(waff1_16, 0, OC, gc_d, OC, nullptr, 0, OC,
       s_aff1, b_aff1, bprime, 1.0f, 1, t64, OC, 0, nullptr, OC, OC, P2);
  gemm(waff2_16, 0, OC, t64, OC, nullptr, 0, OC,
       s_aff2, b_aff2, nullptr, 1.0f, 1, aff_d, OC, 0, nullptr, OC, OC, P2);
  gemm(xc16, 65536, OC, aff_d, OC, nullptr, 0, OC,
       nullptr, nullptr, nullptr, 1.0f, 0, xup_d, OC, 0, nullptr, OC, OC, P2);
  gemm(wup16, 0, 512, xup_d, OC, gc_d, OC, 256,
       s_up, b_up, nullptr, 1.0f, 1, nullptr, 0, 0, out + out_half, OC, 512, P2);
}
